// MultiHeadAttention_64244120814033
// MI455X (gfx1250) — compile-verified, matrix operand order checked by reading, not run on hardware
//
#include <hip/hip_runtime.h>

typedef __attribute__((ext_vector_type(16))) _Float16 v16h;
typedef __attribute__((ext_vector_type(8)))  _Float16 v8h;
typedef __attribute__((ext_vector_type(8)))  float    v8f;
typedef __attribute__((ext_vector_type(4)))  float    v4f;
typedef __attribute__((ext_vector_type(4)))  int      v4i;

#define D_MODEL 1024
#define N_HEADS 16
#define DK      64
#define BATCH   2
#define SEQ     2048
#define NROW    (BATCH * SEQ)

#if __has_builtin(__builtin_amdgcn_global_load_async_to_lds_b128) && \
    __has_builtin(__builtin_amdgcn_s_wait_asynccnt)
#define HAVE_ASYNC_LDS 1
#else
#define HAVE_ASYNC_LDS 0
#endif

typedef __attribute__((address_space(1))) v4i g_v4i;
typedef __attribute__((address_space(3))) v4i l_v4i;

// lane-xor swizzle within 16-lane halves (and_mask=0x1f, or=0, xor=mask<<10)
#define SWZ(v, imm) \
  __int_as_float(__builtin_amdgcn_ds_swizzle(__float_as_int(v), (imm)))
#define XOR16_MAX(t)                \
  do {                              \
    t = fmaxf(t, SWZ(t, 0x041F));   \
    t = fmaxf(t, SWZ(t, 0x081F));   \
    t = fmaxf(t, SWZ(t, 0x101F));   \
    t = fmaxf(t, SWZ(t, 0x201F));   \
  } while (0)
#define XOR16_SUM(t)                \
  do {                              \
    t += SWZ(t, 0x041F);            \
    t += SWZ(t, 0x081F);            \
    t += SWZ(t, 0x101F);            \
    t += SWZ(t, 0x201F);            \
  } while (0)

__device__ __forceinline__ v8f wmma_f16(v16h a, v16h b, v8f c) {
  return __builtin_amdgcn_wmma_f32_16x16x32_f16(
      false, a, false, b, (short)0, c, false, false);
}

// 16-bit A/B fragment layout (ISA 7.12.2, 16x32):
//   lane l: row = (l & 15), hi = l>>4
//   element e (0..15): K = (e < 8 ? 0 : 16) + hi*8 + (e & 7)
__device__ __forceinline__ v16h load_frag_f16(const _Float16* __restrict__ base,
                                              int stride, int row, int kbase,
                                              int lane) {
  int hi = (lane >> 4) & 1;
  const _Float16* p0 = base + (size_t)row * stride + kbase + hi * 8;
  union { v16h v; v8h h[2]; } u;
  u.h[0] = *(const v8h*)(p0);
  u.h[1] = *(const v8h*)(p0 + 16);
  return u.v;
}

// ---------------------------------------------------------------------------
// fp32 -> f16 bulk convert (one-time, trivial vs 23.3 TB/s)
// ---------------------------------------------------------------------------
__global__ __launch_bounds__(256) void cvt_f32_f16(const float* __restrict__ in,
                                                   _Float16* __restrict__ out,
                                                   int n8) {
  int i = blockIdx.x * blockDim.x + threadIdx.x;
  if (i >= n8) return;
  v4f a = *(const v4f*)(in + (size_t)i * 8);
  v4f b = *(const v4f*)(in + (size_t)i * 8 + 4);
  v8h o;
#pragma unroll
  for (int j = 0; j < 4; ++j) {
    o[j]     = (_Float16)a[j];
    o[j + 4] = (_Float16)b[j];
  }
  *(v8h*)(out + (size_t)i * 8) = o;
}

// ---------------------------------------------------------------------------
// Projection GEMM (all-f16 operands, f32 accumulate), 32x64 tile per wave.
// MODE 0: f16 [b,h,s,dk]   MODE 1: f16 [b,h,dk,s]   MODE 2: f32 [m][n]
// ---------------------------------------------------------------------------
template <int MODE>
__global__ __launch_bounds__(128) void proj_gemm(
    const _Float16* __restrict__ X, const _Float16* __restrict__ W,
    void* __restrict__ out, float scale) {
  int lane = threadIdx.x & 31;
  int wave = blockIdx.x * (blockDim.x >> 5) + (threadIdx.x >> 5);
  int mt = wave >> 4;
  int nt = wave & 15;
  int m0 = mt * 32;
  int n0 = nt * 64;
  int hi = lane >> 4;
  int ln = lane & 15;

  v8f acc[2][4] = {};
  for (int kk = 0; kk < D_MODEL; kk += 32) {
    int kp = (kk + 32) & (D_MODEL - 1);   // branchless wrap: always in-bounds
    __builtin_prefetch(X + (size_t)(m0 + ln) * D_MODEL + kp, 0, 0);
    __builtin_prefetch(W + (size_t)(n0 + ln) * D_MODEL + kp, 0, 0);
    v16h a0 = load_frag_f16(X, D_MODEL, m0 + ln, kk, lane);
    v16h a1 = load_frag_f16(X, D_MODEL, m0 + 16 + ln, kk, lane);
#pragma unroll
    for (int t = 0; t < 4; ++t) {
      v16h b = load_frag_f16(W, D_MODEL, n0 + t * 16 + ln, kk, lane);
      acc[0][t] = wmma_f16(a0, b, acc[0][t]);
      acc[1][t] = wmma_f16(a1, b, acc[1][t]);
    }
  }

#pragma unroll
  for (int half = 0; half < 2; ++half) {
#pragma unroll
    for (int t = 0; t < 4; ++t) {
#pragma unroll
      for (int r = 0; r < 8; ++r) {
        int m = m0 + half * 16 + r + 8 * hi;
        int n = n0 + t * 16 + ln;
        float v = acc[half][t][r] * scale;
        if constexpr (MODE == 2) {
          ((float*)out)[(size_t)m * D_MODEL + n] = v;
        } else {
          int b = m >> 11;
          int s = m & (SEQ - 1);
          int h = n >> 6;
          int d = n & (DK - 1);
          _Float16* o = (_Float16*)out;
          if constexpr (MODE == 0)
            o[(((size_t)(b * N_HEADS + h) * SEQ) + s) * DK + d] = (_Float16)v;
          else
            o[(((size_t)(b * N_HEADS + h) * DK) + d) * SEQ + s] = (_Float16)v;
        }
      }
    }
  }
}

// ---------------------------------------------------------------------------
// Fused causal attention, one wave per (b, h, 16 query rows).
// Scores are in the exp2 domain (log2(e)/sqrt(dk) folded into Q).
// K tiles double-buffered into LDS with GLOBAL_LOAD_ASYNC_TO_LDS_B128.
// ---------------------------------------------------------------------------
__global__ __launch_bounds__(128) void flash_attn(
    const _Float16* __restrict__ q16, const _Float16* __restrict__ k16,
    const _Float16* __restrict__ vT16, _Float16* __restrict__ ao16) {
  __shared__ _Float16 ldsP[4][16 * 32];
#if HAVE_ASYNC_LDS
  __shared__ _Float16 ldsK[4][2][32 * DK];   // double-buffered 4 KB K tiles
#endif

  int lane  = threadIdx.x & 31;
  int wslot = threadIdx.x >> 5;
  int wave  = blockIdx.x * 4 + wslot;
  int qt = wave & 127;
  int bh = wave >> 7;
  int q0 = qt * 16;
  int hi = lane >> 4;
  int ln = lane & 15;

  const _Float16* Qp = q16 + (size_t)bh * SEQ * DK;
  const _Float16* Kp = k16 + (size_t)bh * SEQ * DK;
  const _Float16* Vt = vT16 + (size_t)bh * DK * SEQ;

  v16h qa0 = load_frag_f16(Qp, DK, q0 + ln, 0, lane);
  v16h qa1 = load_frag_f16(Qp, DK, q0 + ln, 32, lane);

  v8f O[4] = {};
  float rmax[8], rsum[8];
#pragma unroll
  for (int r = 0; r < 8; ++r) { rmax[r] = -3.0e38f; rsum[r] = 0.f; }

  _Float16* P = ldsP[wslot];
  int nkt = (q0 + 16 + 31) >> 5;

#if HAVE_ASYNC_LDS
  // INST_OFFSET is applied to BOTH the global and LDS addresses (ISA 15.18),
  // so all 8 b128 copies share one global-address pair and one LDS VGPR.
  // Offsets must be literal constants -> manual unroll.
#define ASYNC_CP(SRC, DST, OFF) \
  __builtin_amdgcn_global_load_async_to_lds_b128((g_v4i*)(SRC), (l_v4i*)(DST), OFF, 0)
  auto async_ktile = [&](int k0, _Float16* dst) {
    const _Float16* src = Kp + (size_t)k0 * DK + lane * 8;  // 4 KB contiguous
    _Float16* d = dst + lane * 8;
    ASYNC_CP(src, d, 0);
    ASYNC_CP(src, d, 512);
    ASYNC_CP(src, d, 1024);
    ASYNC_CP(src, d, 1536);
    ASYNC_CP(src, d, 2048);
    ASYNC_CP(src, d, 2560);
    ASYNC_CP(src, d, 3072);
    ASYNC_CP(src, d, 3584);
  };
  async_ktile(0, ldsK[wslot][0]);
#endif

  for (int kt = 0; kt < nkt; ++kt) {
    int k0 = kt * 32;
    {  // branchless prefetch of the next V^T stripe (clamps to the current tile at the end of SEQ)
      int kpre = (k0 + 32 < SEQ) ? (k0 + 32) : k0;
      __builtin_prefetch(Vt + (size_t)ln * SEQ + kpre, 0, 0);
    }

#if HAVE_ASYNC_LDS
    if (kt + 1 < nkt) {
      async_ktile(k0 + 32, ldsK[wslot][(kt + 1) & 1]);
      __builtin_amdgcn_s_wait_asynccnt(8);   // previous tile resident
    } else {
      __builtin_amdgcn_s_wait_asynccnt(0);
    }
    const _Float16* Kt = ldsK[wslot][kt & 1];
    const int kstride = DK, krow = 0;
#else
    const _Float16* Kt = Kp;
    const int kstride = DK, krow = k0;
#endif

    // scores: S = Q(16x64) * K^T(64x32) as two 16x16 column tiles
    v8f s0 = {}, s1 = {};
    {
      v16h kb = load_frag_f16(Kt, kstride, krow + ln, 0, lane);
      s0 = wmma_f16(qa0, kb, s0);
      kb = load_frag_f16(Kt, kstride, krow + ln, 32, lane);
      s0 = wmma_f16(qa1, kb, s0);
      kb = load_frag_f16(Kt, kstride, krow + 16 + ln, 0, lane);
      s1 = wmma_f16(qa0, kb, s1);
      kb = load_frag_f16(Kt, kstride, krow + 16 + ln, 32, lane);
      s1 = wmma_f16(qa1, kb, s1);
    }

    if (kt == nkt - 1) {   // only the diagonal tile needs masking
#pragma unroll
      for (int r = 0; r < 8; ++r) {
        int m = q0 + r + 8 * hi;
        if (k0 + ln > m)      s0[r] = -3.0e38f;
        if (k0 + 16 + ln > m) s1[r] = -3.0e38f;
      }
    }

    // online softmax (exp2 domain); row max reduced across 16 lanes,
    // row sum kept as per-lane partials (corr is lane-uniform) -> reduced once
    float corr[8];
#pragma unroll
    for (int r = 0; r < 8; ++r) {
      float t = fmaxf(s0[r], s1[r]);
      XOR16_MAX(t);
      float mn = fmaxf(rmax[r], t);
      corr[r] = exp2f(rmax[r] - mn);
      rmax[r] = mn;
      float p0 = exp2f(s0[r] - mn);
      float p1 = exp2f(s1[r] - mn);
      s0[r] = p0;
      s1[r] = p1;
      rsum[r] = rsum[r] * corr[r] + p0 + p1;
    }
#pragma unroll
    for (int t = 0; t < 4; ++t)
#pragma unroll
      for (int r = 0; r < 8; ++r) O[t][r] *= corr[r];

    // P: C layout -> LDS -> A layout
#pragma unroll
    for (int r = 0; r < 8; ++r) {
      int m = r + 8 * hi;
      P[m * 32 + ln]      = (_Float16)s0[r];
      P[m * 32 + 16 + ln] = (_Float16)s1[r];
    }
    asm volatile("s_wait_dscnt 0" ::: "memory");
    v16h pa = load_frag_f16(P, 32, ln, 0, lane);

    // O += P(16x32) * V(32x64); V^T layout keeps B loads contiguous
#pragma unroll
    for (int t = 0; t < 4; ++t) {
      v16h vb = load_frag_f16(Vt, SEQ, t * 16 + ln, k0, lane);
      O[t] = wmma_f16(pa, vb, O[t]);
    }
    asm volatile("s_wait_dscnt 0" ::: "memory");  // P reuse next iteration
  }

  // epilogue: reduce row sums across lanes once, then normalize and store
  float inv[8];
#pragma unroll
  for (int r = 0; r < 8; ++r) {
    float s = rsum[r];
    XOR16_SUM(s);
    inv[r] = 1.0f / s;
  }
  int b = bh >> 4, h = bh & 15;
#pragma unroll
  for (int t = 0; t < 4; ++t)
#pragma unroll
    for (int r = 0; r < 8; ++r) {
      int m = q0 + r + 8 * hi;
      int d = h * DK + t * 16 + ln;
      ao16[((size_t)b * SEQ + m) * D_MODEL + d] = (_Float16)(O[t][r] * inv[r]);
    }
}

// ---------------------------------------------------------------------------
extern "C" void kernel_launch(void* const* d_in, const int* in_sizes, int n_in,
                              void* d_out, int out_size, void* d_ws,
                              size_t ws_size, hipStream_t stream) {
  (void)in_sizes; (void)n_in; (void)out_size; (void)ws_size;

  const float* Xq = (const float*)d_in[0];
  const float* Xk = (const float*)d_in[1];
  const float* Xv = (const float*)d_in[2];
  // d_in[3] = mask (causal, computed analytically)
  const float* Wq = (const float*)d_in[4];
  const float* Wk = (const float*)d_in[5];
  const float* Wv = (const float*)d_in[6];
  const float* Wo = (const float*)d_in[7];

  const size_t wElems = (size_t)D_MODEL * D_MODEL;
  const size_t xElems = (size_t)NROW * D_MODEL;
  const size_t qkvElems = (size_t)BATCH * N_HEADS * SEQ * DK;

  _Float16* w16q = (_Float16*)d_ws;
  _Float16* w16k = w16q + wElems;
  _Float16* w16v = w16k + wElems;
  _Float16* w16o = w16v + wElems;
  _Float16* x16q = w16o + wElems;
  _Float16* x16k = x16q + xElems;
  _Float16* x16v = x16k + xElems;
  _Float16* q16  = x16v + xElems;
  _Float16* k16  = q16 + qkvElems;
  _Float16* vT16 = k16 + qkvElems;
  _Float16* ao16 = x16q;   // alias: x16q dead after the Q projection

  dim3 blk256(256), blk128(128);
  int wN8 = (int)(wElems / 8), xN8 = (int)(xElems / 8);
  cvt_f32_f16<<<(wN8 + 255) / 256, blk256, 0, stream>>>(Wq, w16q, wN8);
  cvt_f32_f16<<<(wN8 + 255) / 256, blk256, 0, stream>>>(Wk, w16k, wN8);
  cvt_f32_f16<<<(wN8 + 255) / 256, blk256, 0, stream>>>(Wv, w16v, wN8);
  cvt_f32_f16<<<(wN8 + 255) / 256, blk256, 0, stream>>>(Wo, w16o, wN8);
  cvt_f32_f16<<<(xN8 + 255) / 256, blk256, 0, stream>>>(Xq, x16q, xN8);
  cvt_f32_f16<<<(xN8 + 255) / 256, blk256, 0, stream>>>(Xk, x16k, xN8);
  cvt_f32_f16<<<(xN8 + 255) / 256, blk256, 0, stream>>>(Xv, x16v, xN8);

  int gemmBlocks = (NROW / 32) * (D_MODEL / 64) / 4;   // 512
  int attnBlocks = (BATCH * N_HEADS * (SEQ / 16)) / 4; // 1024

  // 1/sqrt(dk) * log2(e): scores move to the exp2 domain
  const float qscale = 0.125f * 1.44269504088896340736f;
  proj_gemm<0><<<gemmBlocks, blk128, 0, stream>>>(x16q, w16q, q16, qscale);
  proj_gemm<0><<<gemmBlocks, blk128, 0, stream>>>(x16k, w16k, k16, 1.0f);
  proj_gemm<1><<<gemmBlocks, blk128, 0, stream>>>(x16v, w16v, vT16, 1.0f);
  flash_attn<<<attnBlocks, blk128, 0, stream>>>(q16, k16, vT16, ao16);
  proj_gemm<2><<<gemmBlocks, blk128, 0, stream>>>(ao16, w16o, d_out, 1.0f);
}
